// TemporalFeatureAggregator_18219251269816
// MI455X (gfx1250) — compile-verified, matrix operand order checked by reading, not run on hardware
//
#include <hip/hip_runtime.h>
#include <hip/hip_bf16.h>
#include <math.h>

typedef _Float16 half_t;
typedef __attribute__((ext_vector_type(16))) _Float16 v16h;
typedef __attribute__((ext_vector_type(8)))  _Float16 v8h;
typedef __attribute__((ext_vector_type(8)))  float    v8f;

#define PP 16
#define NN 128
#define CC 256
#define SS 64
#define HH 64            // HID = C/4
#define KA (CC * 3)      // 768, im2col K for conv-a (ordered k = dk*256 + c)

static __device__ __forceinline__ v16h join8(v8h lo, v8h hi) {
  return __builtin_shufflevector(lo, hi, 0, 1, 2, 3, 4, 5, 6, 7, 8, 9, 10, 11, 12, 13, 14, 15);
}

static __device__ __forceinline__ v8f wmma_f16(v16h a, v16h b, v8f c) {
  return __builtin_amdgcn_wmma_f32_16x16x32_f16(false, a, false, b, (short)0, c,
                                                false, false);
}

// ---------------------------------------------------------------------------
// Kernel 1: h[br][p][n][hid][s] = leaky_relu(conv1d(x, wa, k=3, pad=1))
// GEMM: M=64 (hid), K=768 (k = dk*256 + c), 64 columns (all s of one n).
// Block = 4 waves; wave w owns hid rows 16w..16w+15 and issues 4 WMMAs per
// 32-wide K chunk (one per 16-column tile), sharing its A fragment.
// grid = (N, 2 branches, P)
// ---------------------------------------------------------------------------
__global__ __launch_bounds__(128) void conv_a_kernel(
    const float* __restrict__ x, const float* __restrict__ w31a,
    const float* __restrict__ w33a, half_t* __restrict__ h31,
    half_t* __restrict__ h33) {
  __shared__ __align__(16) half_t lA[64 * 32];  // [hid][kl]
  __shared__ __align__(16) half_t lB[64 * 32];  // [col s][kl]

  const int tid = threadIdx.x;
  const int wv = tid >> 5;
  const int lane = tid & 31;
  const int g = lane >> 4;   // lane group (0: lanes 0-15, 1: lanes 16-31)
  const int lj = lane & 15;
  const int p = blockIdx.z;
  const int br = blockIdx.y;
  const int n = blockIdx.x;

  const float* wa = (br == 0 ? w31a : w33a) + (size_t)p * HH * KA;
  half_t* hb = (br == 0 ? h31 : h33) + (size_t)p * NN * HH * SS;
  const float* xb = x + (size_t)(p * NN + n) * CC * SS;

  v8f acc[4] = {};
  for (int kc = 0; kc < KA; kc += 32) {
    const int dk = kc >> 8;   // tap index 0..2 (constant per chunk)
    const int ck = kc & 255;  // input-channel base of this chunk

    // Stage A chunk (weights): element (hid, kl) = wa[hid][ck+kl][dk].
#pragma unroll
    for (int i = 0; i < 16; ++i) {
      int e = tid + i * 128;  // 0..2047
      int hid = e >> 5, kl = e & 31;
      lA[e] = (half_t)wa[hid * KA + (ck + kl) * 3 + dk];
    }
    // Stage B chunk (im2col of x): 64 cols x 32 k. Index so consecutive
    // threads read consecutive s (coalesced 256B rows of x).
#pragma unroll
    for (int i = 0; i < 16; ++i) {
      int e = tid + i * 128;  // 0..2047
      int kl = e >> 6, j = e & 63;
      int ss = j + dk - 1;
      float v = (ss >= 0 && ss < SS) ? xb[(ck + kl) * SS + ss] : 0.0f;
      lB[j * 32 + kl] = (half_t)v;
    }
    // Prefetch next chunk's x rows into cache (global_prefetch_b8).
    if (kc + 32 < KA) {
      int kn = kc + 32;
      __builtin_prefetch(xb + ((kn & 255) + (tid >> 2)) * SS, 0, 3);
    }
    __syncthreads();

    // A fragment: lane row = 16*wv + lj; halves 0-7 -> k = 8g..8g+7,
    // halves 8-15 -> k = 16+8g..; two aligned ds_load_b128.
    const int arow = (wv << 4) + lj;
    v16h afrag = join8(*(const v8h*)&lA[arow * 32 + g * 8],
                       *(const v8h*)&lA[arow * 32 + 16 + g * 8]);
#pragma unroll
    for (int t = 0; t < 4; ++t) {
      const int col = (t << 4) + lj;
      v16h bfrag = join8(*(const v8h*)&lB[col * 32 + g * 8],
                         *(const v8h*)&lB[col * 32 + 16 + g * 8]);
      acc[t] = wmma_f16(afrag, bfrag, acc[t]);
    }
    __syncthreads();
  }

  // Epilogue: LeakyReLU, store f16 h[n][hid][s].
#pragma unroll
  for (int t = 0; t < 4; ++t) {
#pragma unroll
    for (int r = 0; r < 8; ++r) {
      float v = acc[t][r];
      v = v >= 0.0f ? v : 0.01f * v;
      int hid = (wv << 4) + r + (g << 3);
      int s = (t << 4) + lj;
      hb[((size_t)n * HH + hid) * SS + s] = (half_t)v;
    }
  }
}

// ---------------------------------------------------------------------------
// Kernel 2: logits via two WMMA GEMMs + fused pooling/gating/max-over-S.
// One workgroup = (p, n, 16-channel tile); wave w owns s = 16w..16w+15.
// Weight tiles staged once per block into LDS as f16 (pre-reordered).
// grid = (C/16, N, P)
// ---------------------------------------------------------------------------
__global__ __launch_bounds__(128) void gate_kernel(
    const float* __restrict__ x, const float* __restrict__ w31b,
    const float* __restrict__ w33b, const half_t* __restrict__ h31,
    const half_t* __restrict__ h33, float* __restrict__ out) {
  __shared__ __align__(16) half_t lH31[SS * HH];        // [s][hid]
  __shared__ __align__(16) half_t lH33[(SS + 2) * HH];  // [s+1][hid], zero border
  __shared__ __align__(16) half_t lW31[16 * HH];        // [c][hid]
  __shared__ __align__(16) half_t lW33[16 * 3 * HH];    // [c][k = dk*64+hid]
  __shared__ __align__(16) float lX[16 * SS];           // [c_local][s] fp32
  __shared__ float red[4 * 16];

  const int tid = threadIdx.x;
  const int wv = tid >> 5;
  const int lane = tid & 31;
  const int g = lane >> 4;
  const int lj = lane & 15;
  const int p = blockIdx.z, n = blockIdx.y;
  const int c0 = blockIdx.x << 4;

  const half_t* hb31 = h31 + (size_t)(p * NN + n) * HH * SS;
  const half_t* hb33 = h33 + (size_t)(p * NN + n) * HH * SS;
  const float* xb = x + ((size_t)(p * NN + n) * CC + c0) * SS;
  const float* wb1 = w31b + (size_t)(p * CC + c0) * HH;
  const float* wb3 = w33b + (size_t)(p * CC + c0) * (HH * 3);

  // Stage h (transposed to [s][hid]), weights (f16), and x rows into LDS.
  for (int i = tid; i < SS * HH; i += 128) {
    int hid = i >> 6, s = i & 63;  // global h layout: [hid][s]
    lH31[s * HH + hid] = hb31[i];
    lH33[(s + 1) * HH + hid] = hb33[i];
  }
  if (tid < HH) {
    lH33[tid] = (half_t)0.0f;                  // s = -1 tap row
    lH33[(SS + 1) * HH + tid] = (half_t)0.0f;  // s = 64 tap row
  }
  for (int i = tid; i < 16 * HH; i += 128) lW31[i] = (half_t)wb1[i];
  for (int i = tid; i < 16 * 3 * HH; i += 128) {
    int cl = i / (3 * HH), k = i - cl * (3 * HH);
    int dk = k >> 6, hid = k & 63;  // reorder to k = dk*64 + hid
    lW33[i] = (half_t)wb3[cl * (3 * HH) + hid * 3 + dk];
  }
  for (int i = tid; i < 16 * SS; i += 128) lX[i] = xb[i];
  __syncthreads();

  const int s = (wv << 4) + lj;  // this lane's column (s position)

  v8f acc31 = {}, acc33 = {};

  // GEMM 3x1: M=16 channels, K=64 (hid).
#pragma unroll
  for (int kc = 0; kc < HH; kc += 32) {
    v16h a = join8(*(const v8h*)&lW31[lj * HH + kc + g * 8],
                   *(const v8h*)&lW31[lj * HH + kc + 16 + g * 8]);
    v16h b = join8(*(const v8h*)&lH31[s * HH + kc + g * 8],
                   *(const v8h*)&lH31[s * HH + kc + 16 + g * 8]);
    acc31 = wmma_f16(a, b, acc31);
  }

  // GEMM 3x3: K=192 with k = dk*64 + hid; each 32-chunk has one tap dk and
  // contiguous hid -> aligned b128 LDS reads for both A and B.
#pragma unroll
  for (int kc = 0; kc < 3 * HH; kc += 32) {
    int dk = kc >> 6;    // {0,0,1,1,2,2}
    int hid0 = kc & 63;  // {0,32,...}
    v16h a = join8(*(const v8h*)&lW33[lj * (3 * HH) + kc + g * 8],
                   *(const v8h*)&lW33[lj * (3 * HH) + kc + 16 + g * 8]);
    v16h b = join8(*(const v8h*)&lH33[(s + dk) * HH + hid0 + g * 8],
                   *(const v8h*)&lH33[(s + dk) * HH + hid0 + 16 + g * 8]);
    acc33 = wmma_f16(a, b, acc33);
  }

  // Fused epilogue: pooling features, sigmoid gating, max over S.
#pragma unroll
  for (int r = 0; r < 8; ++r) {
    int cl = r + (g << 3);  // D layout: VGPR r -> row r (+8 for lanes 16-31)
    const float* xr = &lX[cl * SS];
    float sum3 = 0.0f, mx3 = -INFINITY;
#pragma unroll
    for (int d = -1; d <= 1; ++d) {
      int ss = s + d;
      if (ss >= 0 && ss < SS) {
        float v = xr[ss];
        sum3 += v;
        mx3 = fmaxf(mx3, v);
      }
    }
    float sum5 = 0.0f, mx5 = -INFINITY;
#pragma unroll
    for (int d = -2; d <= 2; ++d) {
      int ss = s + d;
      if (ss >= 0 && ss < SS) {
        float v = xr[ss];
        sum5 += v;
        mx5 = fmaxf(mx5, v);
      }
    }
    float f31 = sum3 * (1.0f / 3.0f) + mx3;
    float f33 = sum5 * 0.2f + mx5;
    float g31 = 1.0f / (1.0f + __expf(-acc31[r]));
    float g33 = 1.0f / (1.0f + __expf(-acc33[r]));
    float val = f31 * g31 + f33 * g33;
    // max over the wave's 16 s-columns (XOR tree stays within 16-lane group)
#pragma unroll
    for (int m = 1; m < 16; m <<= 1) val = fmaxf(val, __shfl_xor(val, m, 32));
    if (lj == 0) red[wv * 16 + cl] = val;
  }
  __syncthreads();
  if (tid < 16) {
    float m = fmaxf(fmaxf(red[tid], red[16 + tid]),
                    fmaxf(red[32 + tid], red[48 + tid]));
    out[(size_t)(p * NN + n) * CC + c0 + tid] = m;
  }
}

// ---------------------------------------------------------------------------
extern "C" void kernel_launch(void* const* d_in, const int* in_sizes, int n_in,
                              void* d_out, int out_size, void* d_ws,
                              size_t ws_size, hipStream_t stream) {
  (void)in_sizes; (void)n_in; (void)out_size; (void)ws_size;
  const float* x    = (const float*)d_in[0];
  const float* w31a = (const float*)d_in[1];
  const float* w31b = (const float*)d_in[2];
  const float* w33a = (const float*)d_in[3];
  const float* w33b = (const float*)d_in[4];

  // Workspace: h31, h33 as f16 [P][N][HH][SS] -> 16 MiB each (32 MiB total).
  half_t* h31 = (half_t*)d_ws;
  half_t* h33 = h31 + (size_t)PP * NN * HH * SS;

  dim3 g1(NN, 2, PP);        // (128, 2, 16), 64x64 output tile per block
  conv_a_kernel<<<g1, 128, 0, stream>>>(x, w31a, w33a, h31, h33);

  dim3 g2(CC / 16, NN, PP);  // (16, 128, 16)
  gate_kernel<<<g2, 128, 0, stream>>>(x, w31b, w33b, h31, h33, (float*)d_out);
}
